// VQVAE_90855738180108
// MI455X (gfx1250) — matrix operand order repaired and checked by reading, not compiled, not run on hardware
//
#include <hip/hip_runtime.h>
#include <hip/hip_bf16.h>

typedef __attribute__((ext_vector_type(16))) _Float16 v16h;
typedef __attribute__((ext_vector_type(8)))  float    v8f;

#define FLAG_IN_RELU  1
#define FLAG_OUT_RELU 2

__device__ __forceinline__ int kmapA(int h, int lo) { return (h < 8 ? h : h + 8) + 8 * lo; }
__device__ __forceinline__ int kmapB(int h, int lo) { return (h < 8 ? h : h + 8) + 8 * lo; }

__device__ __forceinline__ v8f wmma_f16(v16h a, v16h b, v8f c) {
  // v_wmma_f32_16x16x32_f16 : D = A(16x32 f16) * B(32x16 f16) + C(16x16 f32)
  return __builtin_amdgcn_wmma_f32_16x16x32_f16(false, a, false, b, (short)0, c, false, false);
}

// A tile: lane holds row M = m0 + (lane&15); channels c0 + kmapA(h, lane>>4).
// CIN is compile time and c0 is a constant after unrolling -> masks fold away and
// the two contiguous 8-float runs vectorize to global_load_b128.
template <int CIN>
__device__ __forceinline__ v16h loadA_t(const float* __restrict__ base, bool valid,
                                        int c0, int lo, bool doRelu) {
  v16h v;
  if (valid) {
#pragma unroll
    for (int h = 0; h < 16; ++h) {
      int c = c0 + kmapA(h, lo);
      float x = (c < CIN) ? base[c] : 0.0f;
      if (doRelu) x = fmaxf(x, 0.0f);
      v[h] = (_Float16)x;
    }
  } else {
#pragma unroll
    for (int h = 0; h < 16; ++h) v[h] = (_Float16)0.0f;
  }
  return v;
}

// ---------------------------------------------------------------------------
// Weight repack: HWIO f32 -> WMMA-lane-ordered f16 B tiles.
// Layout: [tap][cchunk][ntile][lane(32)] each holding v16h (element h = K slot
// kmapB(h, lane>>4), N = ntile*16 + (lane&15)). One thread per v16h.
// ---------------------------------------------------------------------------
__global__ void pack_w_f16(const float* __restrict__ w, v16h* __restrict__ out,
                           int taps, int Cin, int Cout, int cchunks, int ntiles) {
  int i = blockIdx.x * blockDim.x + threadIdx.x;
  int total = taps * cchunks * ntiles * 32;
  if (i >= total) return;
  int lane = i & 31;
  int j = i >> 5;
  int t = j % ntiles; j /= ntiles;
  int cc = j % cchunks;
  int tap = j / cchunks;
  int lo = lane >> 4, ln = lane & 15;
  int co = t * 16 + ln;
  v16h v;
#pragma unroll
  for (int h = 0; h < 16; ++h) {
    int c = cc * 32 + kmapB(h, lo);
    float x = (c < Cin && co < Cout) ? w[((size_t)tap * Cin + c) * Cout + co] : 0.0f;
    v[h] = (_Float16)x;
  }
  out[i] = v;
}

// Codebook repack: emb[512][128] -> [cc(32)][kc(4)][lane(32)] v16h tiles,
// element h = emb[cc*16 + (lane&15)][kc*32 + kmapB(h, lane>>4)].
__global__ void pack_emb_f16(const float* __restrict__ emb, v16h* __restrict__ out) {
  int i = blockIdx.x * blockDim.x + threadIdx.x;
  if (i >= 32 * 4 * 32) return;
  int lane = i & 31;
  int j = i >> 5;
  int kc = j & 3;
  int cc = j >> 2;
  int lo = lane >> 4, ln = lane & 15;
  int n = cc * 16 + ln;
  v16h v;
#pragma unroll
  for (int h = 0; h < 16; ++h)
    v[h] = (_Float16)emb[(size_t)n * 128 + kc * 32 + kmapB(h, lo)];
  out[i] = v;
}

// ---------------------------------------------------------------------------
// Generic implicit-GEMM conv (NHWC f32 in, packed f16 weights), fused
// bias/relu/residual. One wave = 16 output positions x (NTILES*16) channels.
// ---------------------------------------------------------------------------
template <int NTILES, int CIN>
__global__ void conv_wmma(const float* __restrict__ in, const v16h* __restrict__ wp,
                          const float* __restrict__ bias, const float* __restrict__ resid,
                          float* __restrict__ out,
                          int IH, int IW, int KH, int KW, int Cout,
                          int OH, int OW, int stride, int pad,
                          int flags, int mtiles) {
  constexpr int CCH = (CIN + 31) / 32;
  int wave = threadIdx.x >> 5;
  int lane = threadIdx.x & 31;
  int mtile = blockIdx.x * (blockDim.x >> 5) + wave;
  if (mtile >= mtiles) return;  // wave-uniform: EXEC stays all-ones for WMMA
  int lo = lane >> 4, ln = lane & 15;
  long m0 = (long)mtile * 16;

  if (threadIdx.x == 0) __builtin_prefetch(wp, 0, 1);  // global_prefetch_b8 on weights

  long mA = m0 + ln;
  int ow = (int)(mA % OW);
  long tq = mA / OW;
  int oh = (int)(tq % OH);
  int img = (int)(tq / OH);
  int ih0 = oh * stride - pad;
  int iw0 = ow * stride - pad;
  bool inRelu = (flags & FLAG_IN_RELU) != 0;
  bool outRelu = (flags & FLAG_OUT_RELU) != 0;

  v8f acc[NTILES];
#pragma unroll
  for (int t = 0; t < NTILES; ++t) {
    int n = t * 16 + ln;
    float bv = (bias != nullptr && n < Cout) ? bias[n] : 0.0f;
#pragma unroll
    for (int r = 0; r < 8; ++r) acc[t][r] = bv;
  }

  for (int kh = 0; kh < KH; ++kh) {
    int ih = ih0 + kh;
    for (int kw = 0; kw < KW; ++kw) {
      int iw = iw0 + kw;
      bool valid = (ih >= 0) && (ih < IH) && (iw >= 0) && (iw < IW);
      const float* abase = in + (((size_t)img * IH + ih) * IW + iw) * CIN;
      const v16h* wtap = wp + ((size_t)(kh * KW + kw) * CCH) * NTILES * 32;
#pragma unroll
      for (int cc = 0; cc < CCH; ++cc) {
        v16h a = loadA_t<CIN>(abase, valid, cc * 32, lo, inRelu);
#pragma unroll
        for (int t = 0; t < NTILES; ++t) {
          v16h b = wtap[(cc * NTILES + t) * 32 + lane];  // coalesced 32B f16 load
          acc[t] = wmma_f16(a, b, acc[t]);
        }
      }
    }
  }

#pragma unroll
  for (int t = 0; t < NTILES; ++t) {
    int n = t * 16 + ln;
    if (n >= Cout) continue;
#pragma unroll
    for (int r = 0; r < 8; ++r) {
      long m = m0 + r + 8 * lo;
      float v = acc[t][r];
      if (resid != nullptr) v += resid[(size_t)m * Cout + n];
      if (outRelu) v = fmaxf(v, 0.0f);
      out[(size_t)m * Cout + n] = v;
    }
  }
}

// ---------------------------------------------------------------------------
// Transpose conv k=4 s=2 SAME (out = 2*in), parity-decomposed into 2x2-tap
// gather convs so every M-tile is tap-uniform -> WMMA implicit GEMM.
// M index = (((img*2 + ph)*2 + pw)*IH + ho2)*IW + wo2 ; out(ho,wo)=(2*ho2+ph, 2*wo2+pw)
// ---------------------------------------------------------------------------
template <int NTILES, int CIN>
__global__ void convt_wmma(const float* __restrict__ in, const v16h* __restrict__ wp,
                           const float* __restrict__ bias, float* __restrict__ out,
                           int IH, int IW, int Cout, int flags, int mtiles) {
  constexpr int CCH = (CIN + 31) / 32;
  int wave = threadIdx.x >> 5;
  int lane = threadIdx.x & 31;
  int mtile = blockIdx.x * (blockDim.x >> 5) + wave;
  if (mtile >= mtiles) return;
  int lo = lane >> 4, ln = lane & 15;
  long m0 = (long)mtile * 16;
  int OH = 2 * IH, OW = 2 * IW;
  bool outRelu = (flags & FLAG_OUT_RELU) != 0;

  long mA = m0 + ln;
  int wo2 = (int)(mA % IW);
  long tq = mA / IW;
  int ho2 = (int)(tq % IH);
  int t2 = (int)(tq / IH);
  int pw = t2 & 1, ph = (t2 >> 1) & 1, img = t2 >> 2;

  v8f acc[NTILES];
#pragma unroll
  for (int t = 0; t < NTILES; ++t) {
    int n = t * 16 + ln;
    float bv = (bias != nullptr && n < Cout) ? bias[n] : 0.0f;
#pragma unroll
    for (int r = 0; r < 8; ++r) acc[t][r] = bv;
  }

  for (int a = 0; a < 2; ++a) {
    int kh = ph ? (a ? 2 : 0) : (a ? 3 : 1);
    int dh = ph ? (a ? 0 : 1) : (a ? -1 : 0);
    int hi = ho2 + dh;
    for (int bq = 0; bq < 2; ++bq) {
      int kw = pw ? (bq ? 2 : 0) : (bq ? 3 : 1);
      int dw = pw ? (bq ? 0 : 1) : (bq ? -1 : 0);
      int wi = wo2 + dw;
      bool valid = (hi >= 0) && (hi < IH) && (wi >= 0) && (wi < IW);
      const float* abase = in + (((size_t)img * IH + hi) * IW + wi) * CIN;
      const v16h* wtap = wp + ((size_t)(kh * 4 + kw) * CCH) * NTILES * 32;
#pragma unroll
      for (int cc = 0; cc < CCH; ++cc) {
        v16h av = loadA_t<CIN>(abase, valid, cc * 32, lo, false);
#pragma unroll
        for (int t = 0; t < NTILES; ++t) {
          v16h bv = wtap[(cc * NTILES + t) * 32 + lane];
          acc[t] = wmma_f16(av, bv, acc[t]);
        }
      }
    }
  }

#pragma unroll
  for (int t = 0; t < NTILES; ++t) {
    int n = t * 16 + ln;
    if (n >= Cout) continue;
#pragma unroll
    for (int r = 0; r < 8; ++r) {
      long m = m0 + r + 8 * lo;
      int swo2 = (int)(m % IW);
      long sq = m / IW;
      int sho2 = (int)(sq % IH);
      int st2 = (int)(sq / IH);
      int spw = st2 & 1, sph = (st2 >> 1) & 1, simg = st2 >> 2;
      int ho = 2 * sho2 + sph, wo = 2 * swo2 + spw;
      float v = acc[t][r];
      if (outRelu) v = fmaxf(v, 0.0f);
      out[(((size_t)simg * OH + ho) * OW + wo) * Cout + n] = v;
    }
  }
}

// ---------------------------------------------------------------------------
// VQ: per wave, 16 rows of f (D=128) vs all 512 codes via WMMA; argmin of
// ||e||^2 - 2 f.e tracked per lane, reduced across lanes with shfl_xor.
// Codebook comes pre-packed as f16 B tiles.
// ---------------------------------------------------------------------------
__global__ void vq_argmin_wmma(const float* __restrict__ f, const v16h* __restrict__ embP,
                               const float* __restrict__ eNorm, int* __restrict__ idx,
                               int mtiles) {
  int wave = threadIdx.x >> 5;
  int lane = threadIdx.x & 31;
  int mtile = blockIdx.x * (blockDim.x >> 5) + wave;
  if (mtile >= mtiles) return;
  int lo = lane >> 4, ln = lane & 15;
  long m0 = (long)mtile * 16;

  const float* row = f + (size_t)(m0 + ln) * 128;
  v16h a[4];
#pragma unroll
  for (int kc = 0; kc < 4; ++kc) {
    v16h v;
#pragma unroll
    for (int h = 0; h < 16; ++h) v[h] = (_Float16)row[kc * 32 + kmapA(h, lo)];
    a[kc] = v;
  }

  float bd[8];
  int bi[8];
#pragma unroll
  for (int r = 0; r < 8; ++r) { bd[r] = 3.4e38f; bi[r] = 0; }

  for (int cc = 0; cc < 32; ++cc) {  // 32 chunks x 16 codes = 512
    int n = cc * 16 + ln;
    v8f acc;
#pragma unroll
    for (int r = 0; r < 8; ++r) acc[r] = 0.0f;
#pragma unroll
    for (int kc = 0; kc < 4; ++kc) {
      v16h b = embP[(cc * 4 + kc) * 32 + lane];  // coalesced pre-packed f16
      acc = wmma_f16(a[kc], b, acc);
    }
    float en = eNorm[n];
#pragma unroll
    for (int r = 0; r < 8; ++r) {
      float d = en - 2.0f * acc[r];
      if (d < bd[r]) { bd[r] = d; bi[r] = n; }
    }
  }

#pragma unroll
  for (int r = 0; r < 8; ++r) {
#pragma unroll
    for (int s = 1; s < 16; s <<= 1) {  // lanes sharing a row are within a 16-group
      float od = __shfl_xor(bd[r], s, 32);
      int oi = __shfl_xor(bi[r], s, 32);
      if (od < bd[r] || (od == bd[r] && oi < bi[r])) { bd[r] = od; bi[r] = oi; }
    }
    if (ln == 0) idx[m0 + r + 8 * lo] = bi[r];
  }
}

__global__ void codebook_norms(const float* __restrict__ emb, float* __restrict__ eNorm) {
  int i = blockIdx.x * blockDim.x + threadIdx.x;
  if (i < 512) {
    float s = 0.0f;
    for (int d = 0; d < 128; ++d) { float e = emb[(size_t)i * 128 + d]; s += e * e; }
    eNorm[i] = s;
  }
}

__global__ void gather_codes(const float* __restrict__ emb, const int* __restrict__ idx,
                             float* __restrict__ q, long total) {
  long t = (long)blockIdx.x * blockDim.x + threadIdx.x;
  if (t < total) {
    long m = t >> 7;
    int d = (int)(t & 127);
    q[t] = emb[(size_t)idx[m] * 128 + d];
  }
}

static inline void launch_pack(const float* w, v16h* out, int taps, int Cin, int Cout,
                               int cchunks, int ntiles, hipStream_t stream) {
  int total = taps * cchunks * ntiles * 32;
  pack_w_f16<<<(total + 255) / 256, 256, 0, stream>>>(w, out, taps, Cin, Cout, cchunks, ntiles);
}

extern "C" void kernel_launch(void* const* d_in, const int* in_sizes, int n_in,
                              void* d_out, int out_size, void* d_ws, size_t ws_size,
                              hipStream_t stream) {
  (void)in_sizes; (void)n_in; (void)out_size; (void)ws_size;
  const float* x      = (const float*)d_in[0];
  const float* emb    = (const float*)d_in[1];
  const float* enc_w1 = (const float*)d_in[2];  const float* enc_b1 = (const float*)d_in[3];
  const float* enc_w2 = (const float*)d_in[4];  const float* enc_b2 = (const float*)d_in[5];
  const float* enc_w3 = (const float*)d_in[6];  const float* enc_b3 = (const float*)d_in[7];
  const float* erb1_w1 = (const float*)d_in[8];  const float* erb1_w2 = (const float*)d_in[9];
  const float* erb2_w1 = (const float*)d_in[10]; const float* erb2_w2 = (const float*)d_in[11];
  const float* dec_w  = (const float*)d_in[12]; const float* dec_b  = (const float*)d_in[13];
  const float* drb1_w1 = (const float*)d_in[14]; const float* drb1_w2 = (const float*)d_in[15];
  const float* drb2_w1 = (const float*)d_in[16]; const float* drb2_w2 = (const float*)d_in[17];
  const float* dt1_w  = (const float*)d_in[18]; const float* dt1_b  = (const float*)d_in[19];
  const float* dt2_w  = (const float*)d_in[20]; const float* dt2_b  = (const float*)d_in[21];

  // d_out = [ y (16,256,256,3) | f (16,64,64,128) | q (16,64,64,128) ]
  const long yN = 16L * 256 * 256 * 3;   // 3,145,728
  const long fN = 16L * 64 * 64 * 128;   // 8,388,608
  float* y_out = (float*)d_out;
  float* f_out = y_out + yN;
  float* q_out = f_out + fN;

  // workspace layout (bytes)
  char* ws = (char*)d_ws;
  float* b0 = (float*)(ws);                 // 16*128*128*64 = 67,108,864 B
  float* b1 = (float*)(ws + 67108864);      // 33,554,432 B
  float* b2 = (float*)(ws + 100663296);     // 33,554,432 B
  float* b3 = (float*)(ws + 134217728);     //  8,388,608 B
  float* eNorm = (float*)(ws + 142606336);  // 2,048 B
  int*   idx   = (int*)(ws + 142608384);    // 262,144 B

  // packed-weight arena (f16 WMMA B-tiles), 256B-aligned slices
  size_t po = 142870528;
  auto alloc_pk = [&](size_t nv16) { v16h* p = (v16h*)(ws + po); po += nv16 * 32; return p; };
  v16h* p_enc1 = alloc_pk(16 * 1 * 4 * 32);   // 4x4, Cin3(pad32), Cout64
  v16h* p_enc2 = alloc_pk(16 * 2 * 8 * 32);   // 4x4, 64, 128
  v16h* p_enc3 = alloc_pk(9 * 4 * 8 * 32);    // 3x3, 128, 128
  v16h* p_e1w1 = alloc_pk(9 * 4 * 2 * 32);    // 3x3, 128, 32
  v16h* p_e1w2 = alloc_pk(1 * 1 * 8 * 32);    // 1x1, 32, 128
  v16h* p_e2w1 = alloc_pk(9 * 4 * 2 * 32);
  v16h* p_e2w2 = alloc_pk(1 * 1 * 8 * 32);
  v16h* p_dec  = alloc_pk(9 * 4 * 8 * 32);
  v16h* p_d1w1 = alloc_pk(9 * 4 * 2 * 32);
  v16h* p_d1w2 = alloc_pk(1 * 1 * 8 * 32);
  v16h* p_d2w1 = alloc_pk(9 * 4 * 2 * 32);
  v16h* p_d2w2 = alloc_pk(1 * 1 * 8 * 32);
  v16h* p_dt1  = alloc_pk(16 * 4 * 4 * 32);   // 4x4, 128, 64
  v16h* p_dt2  = alloc_pk(16 * 2 * 1 * 32);   // 4x4, 64, 3(pad16)
  v16h* p_emb  = alloc_pk(32 * 4 * 32);       // codebook B tiles

  // ---- pack all weights (deterministic, runs every call) ----
  launch_pack(enc_w1, p_enc1, 16, 3, 64, 1, 4, stream);
  launch_pack(enc_w2, p_enc2, 16, 64, 128, 2, 8, stream);
  launch_pack(enc_w3, p_enc3, 9, 128, 128, 4, 8, stream);
  launch_pack(erb1_w1, p_e1w1, 9, 128, 32, 4, 2, stream);
  launch_pack(erb1_w2, p_e1w2, 1, 32, 128, 1, 8, stream);
  launch_pack(erb2_w1, p_e2w1, 9, 128, 32, 4, 2, stream);
  launch_pack(erb2_w2, p_e2w2, 1, 32, 128, 1, 8, stream);
  launch_pack(dec_w,  p_dec,  9, 128, 128, 4, 8, stream);
  launch_pack(drb1_w1, p_d1w1, 9, 128, 32, 4, 2, stream);
  launch_pack(drb1_w2, p_d1w2, 1, 32, 128, 1, 8, stream);
  launch_pack(drb2_w1, p_d2w1, 9, 128, 32, 4, 2, stream);
  launch_pack(drb2_w2, p_d2w2, 1, 32, 128, 1, 8, stream);
  launch_pack(dt1_w, p_dt1, 16, 128, 64, 4, 4, stream);
  launch_pack(dt2_w, p_dt2, 16, 64, 3, 2, 1, stream);
  pack_emb_f16<<<16, 256, 0, stream>>>(emb, p_emb);
  codebook_norms<<<2, 256, 0, stream>>>(emb, eNorm);

  const dim3 blk(128);  // 4 waves/block

  // ---- Encoder ----
  conv_wmma<4, 3><<<4096, blk, 0, stream>>>(x, p_enc1, enc_b1, nullptr, b0,
                                            256, 256, 4, 4, 64, 128, 128, 2, 1,
                                            FLAG_OUT_RELU, 16384);
  conv_wmma<8, 64><<<1024, blk, 0, stream>>>(b0, p_enc2, enc_b2, nullptr, b1,
                                             128, 128, 4, 4, 128, 64, 64, 2, 1,
                                             FLAG_OUT_RELU, 4096);
  conv_wmma<8, 128><<<1024, blk, 0, stream>>>(b1, p_enc3, enc_b3, nullptr, b2,
                                              64, 64, 3, 3, 128, 64, 64, 1, 1,
                                              0, 4096);
  conv_wmma<2, 128><<<1024, blk, 0, stream>>>(b2, p_e1w1, nullptr, nullptr, b3,
                                              64, 64, 3, 3, 32, 64, 64, 1, 1,
                                              FLAG_IN_RELU, 4096);
  conv_wmma<8, 32><<<1024, blk, 0, stream>>>(b3, p_e1w2, nullptr, b2, b1,
                                             64, 64, 1, 1, 128, 64, 64, 1, 0,
                                             FLAG_IN_RELU, 4096);
  conv_wmma<2, 128><<<1024, blk, 0, stream>>>(b1, p_e2w1, nullptr, nullptr, b3,
                                              64, 64, 3, 3, 32, 64, 64, 1, 1,
                                              FLAG_IN_RELU, 4096);
  conv_wmma<8, 32><<<1024, blk, 0, stream>>>(b3, p_e2w2, nullptr, b1, f_out,
                                             64, 64, 1, 1, 128, 64, 64, 1, 0,
                                             FLAG_IN_RELU | FLAG_OUT_RELU, 4096);

  // ---- VQ ----
  vq_argmin_wmma<<<1024, blk, 0, stream>>>(f_out, p_emb, eNorm, idx, 4096);
  gather_codes<<<(unsigned)((fN + 255) / 256), 256, 0, stream>>>(emb, idx, q_out, fN);

  // ---- Decoder ----
  conv_wmma<8, 128><<<1024, blk, 0, stream>>>(q_out, p_dec, dec_b, nullptr, b1,
                                              64, 64, 3, 3, 128, 64, 64, 1, 1,
                                              0, 4096);
  conv_wmma<2, 128><<<1024, blk, 0, stream>>>(b1, p_d1w1, nullptr, nullptr, b3,
                                              64, 64, 3, 3, 32, 64, 64, 1, 1,
                                              FLAG_IN_RELU, 4096);
  conv_wmma<8, 32><<<1024, blk, 0, stream>>>(b3, p_d1w2, nullptr, b1, b2,
                                             64, 64, 1, 1, 128, 64, 64, 1, 0,
                                             FLAG_IN_RELU, 4096);
  conv_wmma<2, 128><<<1024, blk, 0, stream>>>(b2, p_d2w1, nullptr, nullptr, b3,
                                              64, 64, 3, 3, 32, 64, 64, 1, 1,
                                              FLAG_IN_RELU, 4096);
  conv_wmma<8, 32><<<1024, blk, 0, stream>>>(b3, p_d2w2, nullptr, b2, b1,
                                             64, 64, 1, 1, 128, 64, 64, 1, 0,
                                             FLAG_IN_RELU | FLAG_OUT_RELU, 4096);
  convt_wmma<4, 128><<<4096, blk, 0, stream>>>(b1, p_dt1, dt1_b, b0,
                                               64, 64, 64, FLAG_OUT_RELU, 16384);
  convt_wmma<1, 64><<<16384, blk, 0, stream>>>(b0, p_dt2, dt2_b, y_out,
                                               128, 128, 3, 0, 65536);
}
